// GATForMultipleChoice_37555194036523
// MI455X (gfx1250) — matrix operand order repaired and checked by reading, not compiled, not run on hardware
//
#include <hip/hip_runtime.h>
#include <hip/hip_bf16.h>
#include <math.h>

// ---------------------------------------------------------------------------
// GAT-for-multiple-choice on MI455X (gfx1250, wave32).
//   Layer-1 GEMM (x @ W1, [10000x300]x[300x512]) -> v_wmma_f32_16x16x32_bf16,
//     register-blocked 16x64 per wave (one head per wave), fused al_src/al_dst
//     epilogue via 16-lane shuffle reductions.
//   Edge softmax / scatter passes -> L2-resident float atomics.
// ---------------------------------------------------------------------------

typedef __attribute__((ext_vector_type(16))) __bf16 v16bf;
typedef __attribute__((ext_vector_type(8)))  __bf16 v8bf;
typedef __attribute__((ext_vector_type(8)))  float  v8f;

#define GAT_N   10000
#define GAT_E   160000
#define GAT_ET  170000      // E + N self-loops
#define GAT_D   300
#define GAT_DP  320         // D padded to multiple of 32 (bf16 WMMA K)
#define GAT_H   8
#define GAT_F1  64
#define GAT_HF  512         // H * F1
#define GAT_A   5
#define SLOPE   0.2f

__device__ __forceinline__ float lrelu(float v) { return v > 0.f ? v : SLOPE * v; }

// Monotone float -> uint key so we can segment-max with integer atomicMax.
__device__ __forceinline__ unsigned fkey(float f) {
    unsigned u = __float_as_uint(f);
    return (u & 0x80000000u) ? ~u : (u | 0x80000000u);
}
__device__ __forceinline__ float funkey(unsigned k) {
    unsigned u = (k & 0x80000000u) ? (k & 0x7fffffffu) : ~k;
    return __uint_as_float(u);
}

// src/dst of concatenated edge list: [ei[0] (E), loops (N)], [ei[1] (E), loops]
__device__ __forceinline__ void edge_sd(const int* __restrict__ ei, int t, int& s, int& d) {
    if (t < GAT_E) { s = ei[t]; d = ei[GAT_E + t]; }
    else           { s = t - GAT_E; d = s; }
}

// ---------------------------------------------------------------------------

__global__ void fill_u32_kernel(unsigned* __restrict__ p, unsigned v, int n) {
    int i = blockIdx.x * blockDim.x + threadIdx.x;
    if (i < n) p[i] = v;
}

// W1 [300x512] f32 -> packed bf16 [(c*512 + col)*32 + kk], K zero-padded to 320.
// This makes each lane's B fragment (16 contiguous K values of one column)
// a pair of contiguous B128 loads in the GEMM.
__global__ void pack_w1_kernel(const float* __restrict__ W1, __bf16* __restrict__ w1pk) {
    int i = blockIdx.x * blockDim.x + threadIdx.x;
    if (i >= (GAT_DP / 32) * GAT_HF * 32) return;
    int kk = i & 31;
    int cj = i >> 5;                // c*512 + col
    int j  = cj & (GAT_HF - 1);
    int c  = cj >> 9;
    int K  = c * 32 + kk;
    float v = (K < GAT_D) ? W1[(size_t)K * GAT_HF + j] : 0.f;
    w1pk[i] = (__bf16)v;
}

// x [N x 300] f32 -> row-major bf16 [N x 320], K zero-padded.
__global__ void convert_x_kernel(const float* __restrict__ x, __bf16* __restrict__ xbf) {
    int i = blockIdx.x * blockDim.x + threadIdx.x;
    if (i >= GAT_N * GAT_DP) return;
    int k = i % GAT_DP;
    int n = i / GAT_DP;
    xbf[i] = (k < GAT_D) ? (__bf16)x[(size_t)n * GAT_D + k] : (__bf16)0.0f;
}

// xp = xbf @ W1 via v_wmma_f32_16x16x32_bf16, register-blocked:
// one wave computes a 16x64 strip (rows m0..m0+15, head hh => cols hh*64..+63),
// reusing one A fragment across 4 WMMAs per K chunk. 625*8 = 5000 wave-strips.
//
// A fragment: lane<16 row M=lane, K = 0..15;
// lane>=16 row M=lane-16, K = 16..31  -> two B128 loads from row-major xbf.
// B fragment: lane col = lane&15, K = kshift + 0..15 (kshift = lane<16 ? 0 : 16)
// -> two B128 loads from packed w1pk.
//
// Epilogue: because the strip spans exactly one head's 64 features, this wave
// owns everything needed for al_src[n,hh] / al_dst[n,hh]; reduce across the
// 16-lane half-wave with shfl_xor and store directly (race-free, no atomics).
__global__ __launch_bounds__(256) void gemm1_wmma_kernel(const __bf16* __restrict__ xbf,
                                                         const __bf16* __restrict__ w1pk,
                                                         const float* __restrict__ a_src,
                                                         const float* __restrict__ a_dst,
                                                         float* __restrict__ xp,
                                                         float* __restrict__ als,
                                                         float* __restrict__ ald) {
    const int lane = threadIdx.x & 31;
    const int wid  = threadIdx.x >> 5;
    const int gt   = blockIdx.x * 8 + wid;    // strip id, exactly 5000
    const int tm   = gt >> 3;                 // 625 row tiles
    const int hh   = gt & 7;                  // head index == 64-col supertile
    const int m0   = tm * 16;
    const int n0   = hh * GAT_F1;
    const int l15  = lane & 15;
    const int aoff = (lane < 16) ? 0 : 16;
    const int boff = (lane < 16) ? 0 : 16;

    v8f acc[4] = {{}, {}, {}, {}};
    const __bf16* arow  = xbf + (size_t)(m0 + l15) * GAT_DP + aoff;
    const __bf16* bbase = w1pk + (size_t)(n0 + l15) * 32 + boff;  // +j*16 cols = +j*512 elems

#pragma unroll
    for (int c = 0; c < GAT_DP / 32; ++c) {
        __builtin_prefetch(arow + 32, 0, 3);   // global_prefetch_b8 next K chunk
        v8bf a0 = *(const v8bf*)(arow);
        v8bf a1 = *(const v8bf*)(arow + 8);
        v16bf A;
#pragma unroll
        for (int i = 0; i < 8; ++i) { A[i] = a0[i]; A[8 + i] = a1[i]; }
#pragma unroll
        for (int j = 0; j < 4; ++j) {
            v8bf b0 = *(const v8bf*)(bbase + j * 512);
            v8bf b1 = *(const v8bf*)(bbase + j * 512 + 8);
            v16bf B;
#pragma unroll
            for (int i = 0; i < 8; ++i) { B[i] = b0[i]; B[8 + i] = b1[i]; }
            acc[j] = __builtin_amdgcn_wmma_f32_16x16x32_bf16(
                /*neg_a=*/false, A, /*neg_b=*/false, B,
                /*c_mod=*/(short)0, acc[j], /*reuse_a=*/false, /*reuse_b=*/false);
        }
        arow  += 32;
        bbase += (size_t)GAT_HF * 32;
    }

    // C/D layout: VGPR r -> row m0 + r (+8 for lanes 16..31), col = n0+16j+(lane&15)
#pragma unroll
    for (int j = 0; j < 4; ++j) {
#pragma unroll
        for (int r = 0; r < 8; ++r) {
            int row = m0 + r + ((lane < 16) ? 0 : 8);
            xp[(size_t)row * GAT_HF + n0 + 16 * j + l15] = acc[j][r];
        }
    }

    // Fused attention projections for this head.
    float av[4], dv[4];
#pragma unroll
    for (int j = 0; j < 4; ++j) {
        av[j] = a_src[n0 + 16 * j + l15];     // a_src1[hh, 16j + l15]
        dv[j] = a_dst[n0 + 16 * j + l15];
    }
#pragma unroll
    for (int r = 0; r < 8; ++r) {
        float s = 0.f, d = 0.f;
#pragma unroll
        for (int j = 0; j < 4; ++j) { s += acc[j][r] * av[j]; d += acc[j][r] * dv[j]; }
#pragma unroll
        for (int o = 8; o > 0; o >>= 1) {     // reduce within each 16-lane half
            s += __shfl_xor(s, o, 16);
            d += __shfl_xor(d, o, 16);
        }
        if (l15 == 0) {
            int row = m0 + r + ((lane < 16) ? 0 : 8);
            als[row * GAT_H + hh] = s;
            ald[row * GAT_H + hh] = d;
        }
    }
}

__global__ void edge_max1_kernel(const int* __restrict__ ei, const float* __restrict__ als,
                                 const float* __restrict__ ald, unsigned* __restrict__ emaxk) {
    int t = blockIdx.x * blockDim.x + threadIdx.x;
    if (t >= GAT_ET) return;
    int s, d; edge_sd(ei, t, s, d);
#pragma unroll
    for (int h = 0; h < GAT_H; ++h) {
        float v = lrelu(als[s * GAT_H + h] + ald[d * GAT_H + h]);
        atomicMax(&emaxk[d * GAT_H + h], fkey(v));
    }
}

__global__ void edge_denom1_kernel(const int* __restrict__ ei, const float* __restrict__ als,
                                   const float* __restrict__ ald,
                                   const unsigned* __restrict__ emaxk, float* __restrict__ den) {
    int t = blockIdx.x * blockDim.x + threadIdx.x;
    if (t >= GAT_ET) return;
    int s, d; edge_sd(ei, t, s, d);
#pragma unroll
    for (int h = 0; h < GAT_H; ++h) {
        float v = lrelu(als[s * GAT_H + h] + ald[d * GAT_H + h]);
        atomicAdd(&den[d * GAT_H + h], __expf(v - funkey(emaxk[d * GAT_H + h])));
    }
}

// out1[dst, h*64+f] += alpha * xp[src, h*64+f]  (one thread per (edge, head))
__global__ void edge_agg1_kernel(const int* __restrict__ ei, const float* __restrict__ als,
                                 const float* __restrict__ ald,
                                 const unsigned* __restrict__ emaxk, const float* __restrict__ den,
                                 const float* __restrict__ xp, float* __restrict__ out1) {
    int i = blockIdx.x * blockDim.x + threadIdx.x;
    if (i >= GAT_ET * GAT_H) return;
    int h = i & 7, t = i >> 3;
    int s, d; edge_sd(ei, t, s, d);
    float v = lrelu(als[s * GAT_H + h] + ald[d * GAT_H + h]);
    float alpha = __expf(v - funkey(emaxk[d * GAT_H + h])) / den[d * GAT_H + h];
    const float* xs = xp + (size_t)s * GAT_HF + h * GAT_F1;
    float* od = out1 + (size_t)d * GAT_HF + h * GAT_F1;
#pragma unroll 4
    for (int f = 0; f < GAT_F1; ++f) atomicAdd(&od[f], alpha * xs[f]);
}

// h1 = relu(out1 + b1) in place; xp2[n] = <h1[n,:], W2>. One wave per node.
__global__ __launch_bounds__(256) void h1_xp2_kernel(float* __restrict__ h1,
                                                     const float* __restrict__ b1,
                                                     const float* __restrict__ W2,
                                                     float* __restrict__ xp2) {
    int lane = threadIdx.x & 31, wid = threadIdx.x >> 5;
    int n = blockIdx.x * 8 + wid;
    if (n >= GAT_N) return;
    float* row = h1 + (size_t)n * GAT_HF;
    float s = 0.f;
#pragma unroll
    for (int f = lane; f < GAT_HF; f += 32) {
        float v = row[f] + b1[f];
        v = v > 0.f ? v : 0.f;
        row[f] = v;
        s += v * W2[f];
    }
#pragma unroll
    for (int o = 16; o > 0; o >>= 1) s += __shfl_down(s, o, 32);
    if (lane == 0) xp2[n] = s;
}

__global__ void edge_max2_kernel(const int* __restrict__ ei, const float* __restrict__ xp2,
                                 const float* __restrict__ as2p, const float* __restrict__ ad2p,
                                 unsigned* __restrict__ emx2, int* __restrict__ cnt) {
    int t = blockIdx.x * blockDim.x + threadIdx.x;
    if (t >= GAT_ET) return;
    int s, d; edge_sd(ei, t, s, d);
    float v = lrelu(xp2[s] * as2p[0] + xp2[d] * ad2p[0]);
    atomicMax(&emx2[d], fkey(v));
    atomicAdd(&cnt[d], 1);
}

__global__ void edge_denom2_kernel(const int* __restrict__ ei, const float* __restrict__ xp2,
                                   const float* __restrict__ as2p, const float* __restrict__ ad2p,
                                   const unsigned* __restrict__ emx2, float* __restrict__ den2) {
    int t = blockIdx.x * blockDim.x + threadIdx.x;
    if (t >= GAT_ET) return;
    int s, d; edge_sd(ei, t, s, d);
    float v = lrelu(xp2[s] * as2p[0] + xp2[d] * ad2p[0]);
    atomicAdd(&den2[d], __expf(v - funkey(emx2[d])));
}

// out2[dst] += alpha2 * xp2[src]; attn_row[t] = counts[dst] * alpha2 for first N edges
__global__ void edge_agg2_kernel(const int* __restrict__ ei, const float* __restrict__ xp2,
                                 const float* __restrict__ as2p, const float* __restrict__ ad2p,
                                 const unsigned* __restrict__ emx2, const float* __restrict__ den2,
                                 const int* __restrict__ cnt, float* __restrict__ out2,
                                 float* __restrict__ attn_row) {
    int t = blockIdx.x * blockDim.x + threadIdx.x;
    if (t >= GAT_ET) return;
    int s, d; edge_sd(ei, t, s, d);
    float v = lrelu(xp2[s] * as2p[0] + xp2[d] * ad2p[0]);
    float alpha = __expf(v - funkey(emx2[d])) / den2[d];
    atomicAdd(&out2[d], alpha * xp2[s]);
    if (t < GAT_N) attn_row[t] = (float)cnt[d] * alpha;     // matches alpha2[:N, 0]
}

__global__ void proba_reduce_kernel(const float* __restrict__ out2, float* __restrict__ slot) {
    float s = 0.f;
    for (int i = blockIdx.x * blockDim.x + threadIdx.x; i < GAT_N; i += gridDim.x * blockDim.x)
        s += out2[i];
#pragma unroll
    for (int o = 16; o > 0; o >>= 1) s += __shfl_down(s, o, 32);
    if ((threadIdx.x & 31) == 0) atomicAdd(slot, s);
}

// out[0..4] = log_softmax(proba), out[5..] = attn[argmax]
__global__ void finalize_kernel(const float* __restrict__ proba, const float* __restrict__ b2,
                                const float* __restrict__ attn, float* __restrict__ out) {
    int t = blockIdx.x * blockDim.x + threadIdx.x;
    if (t >= GAT_A + GAT_N) return;
    float p[GAT_A];
    float m = -1e30f;
#pragma unroll
    for (int i = 0; i < GAT_A; ++i) {
        p[i] = proba[i] * (1.0f / (float)GAT_N) + b2[0];
        m = fmaxf(m, p[i]);
    }
    float sum = 0.f;
#pragma unroll
    for (int i = 0; i < GAT_A; ++i) sum += __expf(p[i] - m);
    float lse = m + __logf(sum);
    int best = 0;
#pragma unroll
    for (int i = 1; i < GAT_A; ++i) if (p[i] > p[best]) best = i;
    if (t < GAT_A) out[t] = p[t] - lse;
    else           out[t] = attn[(size_t)best * GAT_N + (t - GAT_A)];
}

// ---------------------------------------------------------------------------

extern "C" void kernel_launch(void* const* d_in, const int* in_sizes, int n_in,
                              void* d_out, int out_size, void* d_ws, size_t ws_size,
                              hipStream_t stream) {
    const float* x   = (const float*)d_in[0];
    const int*   eix = (const int*)d_in[1];
    const float* W1  = (const float*)d_in[2];
    const float* as1 = (const float*)d_in[3];
    const float* ad1 = (const float*)d_in[4];
    const float* b1  = (const float*)d_in[5];
    const float* W2  = (const float*)d_in[6];
    const float* as2 = (const float*)d_in[7];
    const float* ad2 = (const float*)d_in[8];
    const float* b2  = (const float*)d_in[9];
    float* out = (float*)d_out;

    // ---- workspace carving (256B aligned) ----
    char* w = (char*)d_ws;
    size_t off = 0;
    auto alloc = [&](size_t bytes) {
        off = (off + 255) & ~(size_t)255;
        size_t o = off; off += bytes; return o;
    };
    size_t o_w1pk = alloc((size_t)GAT_DP * GAT_HF * 2);
    size_t o_xbf  = alloc((size_t)GAT_N * GAT_DP * 2);
    size_t o_xp   = alloc((size_t)GAT_N * GAT_HF * 4);
    size_t o_als  = alloc((size_t)GAT_N * GAT_H * 4);
    size_t o_ald  = alloc((size_t)GAT_N * GAT_H * 4);
    size_t o_emax = alloc((size_t)GAT_N * GAT_H * 4);
    size_t o_den  = alloc((size_t)GAT_N * GAT_H * 4);
    size_t o_out1 = alloc((size_t)GAT_N * GAT_HF * 4);
    size_t o_xp2  = alloc((size_t)GAT_N * 4);
    size_t o_emx2 = alloc((size_t)GAT_N * 4);
    size_t o_den2 = alloc((size_t)GAT_N * 4);
    size_t o_out2 = alloc((size_t)GAT_N * 4);
    size_t o_cnt  = alloc((size_t)GAT_N * 4);
    size_t o_prob = alloc((size_t)GAT_A * 4);
    size_t o_attn = alloc((size_t)GAT_A * GAT_N * 4);
    (void)ws_size;

    __bf16*   w1pk = (__bf16*)(w + o_w1pk);
    __bf16*   xbf  = (__bf16*)(w + o_xbf);
    float*    xp   = (float*)(w + o_xp);
    float*    als  = (float*)(w + o_als);
    float*    ald  = (float*)(w + o_ald);
    unsigned* emax = (unsigned*)(w + o_emax);
    float*    den  = (float*)(w + o_den);
    float*    out1 = (float*)(w + o_out1);
    float*    xp2  = (float*)(w + o_xp2);
    unsigned* emx2 = (unsigned*)(w + o_emx2);
    float*    den2 = (float*)(w + o_den2);
    float*    out2 = (float*)(w + o_out2);
    int*      cnt  = (int*)(w + o_cnt);
    float*    prob = (float*)(w + o_prob);
    float*    attn = (float*)(w + o_attn);

    const int T = 256;
    auto blk = [](int n, int t) { return (n + t - 1) / t; };

    fill_u32_kernel<<<1, 32, 0, stream>>>((unsigned*)prob, 0u, GAT_A);
    pack_w1_kernel<<<blk((GAT_DP / 32) * GAT_HF * 32, T), T, 0, stream>>>(W1, w1pk);

    for (int a = 0; a < GAT_A; ++a) {
        const float* xa = x + (size_t)a * GAT_N * GAT_D;
        const int*   ea = eix + (size_t)a * 2 * GAT_E;
        float* attn_row = attn + (size_t)a * GAT_N;

        // ---- layer 1 ----
        convert_x_kernel<<<blk(GAT_N * GAT_DP, T), T, 0, stream>>>(xa, xbf);
        gemm1_wmma_kernel<<<625, 256, 0, stream>>>(xbf, w1pk, as1, ad1, xp, als, ald);

        fill_u32_kernel<<<blk(GAT_N * GAT_H, T), T, 0, stream>>>(emax, 0u, GAT_N * GAT_H);
        fill_u32_kernel<<<blk(GAT_N * GAT_H, T), T, 0, stream>>>((unsigned*)den, 0u, GAT_N * GAT_H);
        fill_u32_kernel<<<blk(GAT_N * GAT_HF, T), T, 0, stream>>>((unsigned*)out1, 0u, GAT_N * GAT_HF);

        edge_max1_kernel<<<blk(GAT_ET, T), T, 0, stream>>>(ea, als, ald, emax);
        edge_denom1_kernel<<<blk(GAT_ET, T), T, 0, stream>>>(ea, als, ald, emax, den);
        edge_agg1_kernel<<<blk(GAT_ET * GAT_H, T), T, 0, stream>>>(ea, als, ald, emax, den, xp, out1);

        // ---- layer 2 ----
        h1_xp2_kernel<<<GAT_N / 8, 256, 0, stream>>>(out1, b1, W2, xp2);

        fill_u32_kernel<<<blk(GAT_N, T), T, 0, stream>>>(emx2, 0u, GAT_N);
        fill_u32_kernel<<<blk(GAT_N, T), T, 0, stream>>>((unsigned*)den2, 0u, GAT_N);
        fill_u32_kernel<<<blk(GAT_N, T), T, 0, stream>>>((unsigned*)out2, 0u, GAT_N);
        fill_u32_kernel<<<blk(GAT_N, T), T, 0, stream>>>((unsigned*)cnt, 0u, GAT_N);

        edge_max2_kernel<<<blk(GAT_ET, T), T, 0, stream>>>(ea, xp2, as2, ad2, emx2, cnt);
        edge_denom2_kernel<<<blk(GAT_ET, T), T, 0, stream>>>(ea, xp2, as2, ad2, emx2, den2);
        edge_agg2_kernel<<<blk(GAT_ET, T), T, 0, stream>>>(ea, xp2, as2, ad2, emx2, den2, cnt,
                                                           out2, attn_row);

        proba_reduce_kernel<<<40, 256, 0, stream>>>(out2, prob + a);
    }

    finalize_kernel<<<blk(GAT_A + GAT_N, T), T, 0, stream>>>(prob, b2, attn, out);
}
